// GNNLayer_light_31284541784161
// MI455X (gfx1250) — compile-verified, matrix operand order checked by reading, not run on hardware
//
#include <hip/hip_runtime.h>
#include <hip/hip_bf16.h>

// ---------------------------------------------------------------------------
// GNN gated GraphConv layer, fused for MI455X (gfx1250, wave32, WMMA).
// B=2, Vsc=Vst=256, H=256.
//   e_new = e@C_w + C_b + Ah + Bh ; gates = sigmoid(e_new)
//   h1_new = Uh1 + sum_t gates*V2h ; h2_new = Uh2 + sum_s gates*V1h
//   outs   = residual + relu(LayerNorm(...))
// GEMMs use bf16 hi/lo split (3x v_wmma_f32_16x16x32_bf16, f32 accum) for
// ~2^-16 relative accuracy at matrix-core speed.
// ---------------------------------------------------------------------------

typedef __attribute__((ext_vector_type(8)))  float  v8f;
typedef __attribute__((ext_vector_type(16))) __bf16 v16bf;

#define Hh 256
#define ROWS_H 131072  // B*V*H elements of each (B,V,H) tensor

static __device__ __forceinline__ unsigned short bfhi(float f) {
  unsigned int u = __float_as_uint(f);
  return (unsigned short)((u + 0x7FFFu + ((u >> 16) & 1u)) >> 16);
}
static __device__ __forceinline__ float bf2f(unsigned short s) {
  return __uint_as_float(((unsigned int)s) << 16);
}

static __device__ __forceinline__ v8f wmma_bf16(v16bf a, v16bf b, v8f c) {
  return __builtin_amdgcn_wmma_f32_16x16x32_bf16(false, a, false, b, (short)0, c,
                                                 false, false);
}

// Load 16 consecutive f32 and split into bf16 hi/lo fragments (A-matrix:
// lane L holds row M=L%16, K = (L/16)*16 + i, packed pairs -> v16bf).
static __device__ __forceinline__ void splitA16(const float* __restrict__ p,
                                                v16bf& hi, v16bf& lo) {
  const float4* q = (const float4*)p;
  float4 a = q[0], b = q[1], c = q[2], d = q[3];
  float f[16] = {a.x, a.y, a.z, a.w, b.x, b.y, b.z, b.w,
                 c.x, c.y, c.z, c.w, d.x, d.y, d.z, d.w};
  union { v16bf v; unsigned short u[16]; } H, L;
  #pragma unroll
  for (int i = 0; i < 16; ++i) {
    unsigned short h = bfhi(f[i]);
    H.u[i] = h;
    L.u[i] = bfhi(f[i] - bf2f(h));
  }
  hi = H.v;
  lo = L.v;
}

// Shared WMMA core: one wave computes a 16(M) x 256(N) strip with K=256.
// ArowLane: per-lane pointer to its A row (lane M = lane&15).
// kSel = (lane>>4)*16 ; ln = lane&15.
// B is pre-transposed bf16 [n][k] so each lane loads 32 contiguous bytes.
static __device__ __forceinline__ void gemm_core(
    const float* __restrict__ ArowLane, int kSel, int ln,
    const unsigned short* __restrict__ bHi,
    const unsigned short* __restrict__ bLo, v8f* __restrict__ acc) {
  for (int kb = 0; kb < 256; kb += 32) {
    int ka = kb + kSel;
    v16bf ahi, alo;
    splitA16(ArowLane + ka, ahi, alo);
    #pragma unroll
    for (int nt = 0; nt < 16; ++nt) {
      int off = (nt * 16 + ln) * 256 + ka;  // bf16 elements
      union { uint4 q[2]; v16bf v; } BH, BL;
      const uint4* ph = (const uint4*)(bHi + off);
      BH.q[0] = ph[0]; BH.q[1] = ph[1];
      const uint4* pl = (const uint4*)(bLo + off);
      BL.q[0] = pl[0]; BL.q[1] = pl[1];
      acc[nt] = wmma_bf16(ahi, BH.v, acc[nt]);  // hi*hi
      acc[nt] = wmma_bf16(alo, BH.v, acc[nt]);  // lo*hi
      acc[nt] = wmma_bf16(ahi, BL.v, acc[nt]);  // hi*lo
    }
  }
}

// ---------------------------------------------------------------------------
// Kernel 0: transpose+split the 7 weight matrices W[k][n] (f32) into
// bf16 hi/lo stored as [n][k] so B-fragment loads are contiguous.
// grid (4,4,7), block 256. LDS-tiled 64x64 transpose.
// ---------------------------------------------------------------------------
__global__ __launch_bounds__(256) void conv_w(
    const float* __restrict__ w0, const float* __restrict__ w1,
    const float* __restrict__ w2, const float* __restrict__ w3,
    const float* __restrict__ w4, const float* __restrict__ w5,
    const float* __restrict__ w6, unsigned short* __restrict__ wHiT,
    unsigned short* __restrict__ wLoT) {
  __shared__ unsigned short hiT[64][65];
  __shared__ unsigned short loT[64][65];
  int wsel = blockIdx.z;
  const float* W = (wsel == 0) ? w0 : (wsel == 1) ? w1 : (wsel == 2) ? w2
                 : (wsel == 3) ? w3 : (wsel == 4) ? w4 : (wsel == 5) ? w5 : w6;
  int k0 = blockIdx.x * 64, n0 = blockIdx.y * 64;
  int tid = threadIdx.x;
  for (int idx = tid; idx < 4096; idx += 256) {
    int r = idx >> 6, c = idx & 63;  // r=k_local, c=n_local
    float f = W[(size_t)(k0 + r) * 256 + (n0 + c)];
    unsigned short h = bfhi(f);
    hiT[c][r] = h;
    loT[c][r] = bfhi(f - bf2f(h));
  }
  __syncthreads();
  unsigned short* oh = wHiT + (size_t)wsel * 65536;
  unsigned short* ol = wLoT + (size_t)wsel * 65536;
  for (int idx = tid; idx < 4096; idx += 256) {
    int r = idx >> 6, c = idx & 63;  // r=n_local, c=k_local
    oh[(size_t)(n0 + r) * 256 + (k0 + c)] = hiT[r][c];
    ol[(size_t)(n0 + r) * 256 + (k0 + c)] = loT[r][c];
  }
}

// ---------------------------------------------------------------------------
// Kernel 1: six (512x256)@(256x256)+bias projections. grid (4,6), block 256.
// g: 0 U1(h1)->h1_acc, 1 U2(h2)->h2_acc, 2 V1(h1)->V1h, 3 V2(h2)->V2h,
//    4 A(h1)->Ah, 5 Bm(h2)->Bh.   (weights in ws ordered U1,U2,V1,V2,A,Bm,C)
// ---------------------------------------------------------------------------
__global__ __launch_bounds__(256) void small_gemms(
    const float* __restrict__ h1, const float* __restrict__ h2,
    const unsigned short* __restrict__ wHiT,
    const unsigned short* __restrict__ wLoT, const float* __restrict__ U1b,
    const float* __restrict__ U2b, const float* __restrict__ V1b,
    const float* __restrict__ V2b, const float* __restrict__ Ab,
    const float* __restrict__ Bb, float* __restrict__ h1_acc,
    float* __restrict__ h2_acc, float* __restrict__ V1h,
    float* __restrict__ V2h, float* __restrict__ Ah, float* __restrict__ Bh) {
  int g = blockIdx.y;
  const float* X;
  const float* bias;
  float* out;
  switch (g) {
    case 0: X = h1; bias = U1b; out = h1_acc; break;
    case 1: X = h2; bias = U2b; out = h2_acc; break;
    case 2: X = h1; bias = V1b; out = V1h; break;
    case 3: X = h2; bias = V2b; out = V2h; break;
    case 4: X = h1; bias = Ab;  out = Ah;  break;
    default: X = h2; bias = Bb; out = Bh;  break;
  }
  const unsigned short* bHi = wHiT + (size_t)g * 65536;
  const unsigned short* bLo = wLoT + (size_t)g * 65536;
  int tid = threadIdx.x, w = tid >> 5, lane = tid & 31;
  int half = lane >> 4, ln = lane & 15;
  int rowBase = blockIdx.x * 128 + w * 16;
  const float* ArowLane = X + (size_t)(rowBase + ln) * 256;
  v8f acc[16];
  #pragma unroll
  for (int nt = 0; nt < 16; ++nt)
    #pragma unroll
    for (int j = 0; j < 8; ++j) acc[nt][j] = 0.f;
  gemm_core(ArowLane, half * 16, ln, bHi, bLo, acc);
  #pragma unroll
  for (int nt = 0; nt < 16; ++nt) {
    int n = nt * 16 + ln;
    float bb = bias[n];
    #pragma unroll
    for (int i = 0; i < 8; ++i) {
      int M = half * 8 + i;  // D layout: lanes 0-15 rows 0-7, lanes 16-31 rows 8-15
      out[(size_t)(rowBase + M) * 256 + n] = acc[nt][i] + bb;
    }
  }
}

// ---------------------------------------------------------------------------
// Kernel 2: fused e-path. grid 1024 = (b:2, sTile:32, tTile:16), block 256.
// Block covers 8 s x 16 t rows of e. Wave w owns s_local=w, its 16 rows are
// the 16 t's -> WMMA M-rows. Full N=256 kept in 16 accumulator tiles so
// LayerNorm stats are done with in-register shfl_xor butterflies.
// ---------------------------------------------------------------------------
__global__ __launch_bounds__(256) void fused_e(
    const float* __restrict__ e, const unsigned short* __restrict__ cwHi,
    const unsigned short* __restrict__ cwLo, const float* __restrict__ Cb,
    const float* __restrict__ Ah, const float* __restrict__ Bh,
    const float* __restrict__ V1h, const float* __restrict__ V2h,
    const float* __restrict__ ge, const float* __restrict__ be,
    float* __restrict__ h1_acc, float* __restrict__ h2_acc,
    float* __restrict__ e_out) {
  __shared__ float h2part[16 * 256];
  int tid = threadIdx.x, w = tid >> 5, lane = tid & 31;
  int half = lane >> 4, ln = lane & 15;
  int blk = blockIdx.x;
  int tIdx = blk & 15, sIdx = (blk >> 4) & 31, b = blk >> 9;
  int s0 = sIdx * 8, t0 = tIdx * 16;

  for (int i = tid; i < 4096; i += 256) h2part[i] = 0.f;
  __syncthreads();

  int sG = s0 + w;
  size_t hRowS = ((size_t)b * 256 + sG) * 256;   // base of (b,sG,:) in (B,V,H)
  size_t hRowT = ((size_t)b * 256 + t0) * 256;   // base of (b,t0,:) in (B,V,H)
  size_t eBase = (((size_t)b * 256 + sG) * 256 + t0) * 256;  // e[b][sG][t0][0]

  const float* ArowLane = e + eBase + (size_t)ln * 256;  // row M = t_local = ln
  v8f acc[16];
  #pragma unroll
  for (int nt = 0; nt < 16; ++nt)
    #pragma unroll
    for (int j = 0; j < 8; ++j) acc[nt][j] = 0.f;

  gemm_core(ArowLane, half * 16, ln, cwHi, cwLo, acc);

  // ---- epilogue: e_new = Ce + C_b + Ah + Bh, stats, gates, outputs ----
  float cah[16], v1v[16];
  #pragma unroll
  for (int nt = 0; nt < 16; ++nt) {
    int n = nt * 16 + ln;
    cah[nt] = Cb[n] + Ah[hRowS + n];
    v1v[nt] = V1h[hRowS + n];
  }

  float rs[8] = {0, 0, 0, 0, 0, 0, 0, 0};
  float rq[8] = {0, 0, 0, 0, 0, 0, 0, 0};
  #pragma unroll
  for (int nt = 0; nt < 16; ++nt) {
    int n = nt * 16 + ln;
    const float* BhCol = Bh + hRowT + n;
    #pragma unroll
    for (int i = 0; i < 8; ++i) {
      int M = half * 8 + i;
      float x = acc[nt][i] + cah[nt] + BhCol[(size_t)M * 256];
      acc[nt][i] = x;
      rs[i] += x;
      rq[i] += x * x;
    }
  }
  // reduce across the 16 lanes that share each row group
  #pragma unroll
  for (int i = 0; i < 8; ++i) {
    for (int m = 8; m >= 1; m >>= 1) {
      rs[i] += __shfl_xor(rs[i], m, 32);
      rq[i] += __shfl_xor(rq[i], m, 32);
    }
  }
  float mean[8], rstd[8];
  #pragma unroll
  for (int i = 0; i < 8; ++i) {
    mean[i] = rs[i] * (1.f / 256.f);
    float var = rq[i] * (1.f / 256.f) - mean[i] * mean[i];
    rstd[i] = rsqrtf(var + 1e-5f);
  }

  float h1p[16];
  #pragma unroll
  for (int nt = 0; nt < 16; ++nt) h1p[nt] = 0.f;

  #pragma unroll
  for (int nt = 0; nt < 16; ++nt) {
    int n = nt * 16 + ln;
    float gm = ge[n], bt = be[n];
    float vv = v1v[nt];
    const float* V2Col = V2h + hRowT + n;
    const float* eCol = e + eBase + n;
    float* oCol = e_out + eBase + n;
    #pragma unroll
    for (int i = 0; i < 8; ++i) {
      int M = half * 8 + i;
      float x = acc[nt][i];
      float g = 1.0f / (1.0f + __expf(-x));              // sigmoid gate
      h1p[nt] += g * V2Col[(size_t)M * 256];             // sum over t (in-reg)
      atomicAdd(&h2part[M * 256 + n], g * vv);           // sum over s (LDS)
      float y = (x - mean[i]) * rstd[i] * gm + bt;       // LayerNorm
      y = fmaxf(y, 0.f);                                 // relu
      oCol[(size_t)M * 256] = eCol[(size_t)M * 256] + y; // residual
    }
  }

  // fold the two row-halves, then one fp-atomic per (s,h)
  #pragma unroll
  for (int nt = 0; nt < 16; ++nt) h1p[nt] += __shfl_xor(h1p[nt], 16, 32);
  if (half == 0) {
    #pragma unroll
    for (int nt = 0; nt < 16; ++nt)
      unsafeAtomicAdd(&h1_acc[hRowS + nt * 16 + ln], h1p[nt]);
  }

  __syncthreads();
  for (int i = tid; i < 4096; i += 256) {
    int M = i >> 8, n = i & 255;
    unsafeAtomicAdd(&h2_acc[hRowT + (size_t)M * 256 + n], h2part[i]);
  }
}

// ---------------------------------------------------------------------------
// Kernel 3: h1_out/h2_out = h + relu(LayerNorm(h_acc)). 1024 rows, 1/wave.
// ---------------------------------------------------------------------------
__global__ __launch_bounds__(256) void finalize_h(
    const float* __restrict__ h1, const float* __restrict__ h2,
    const float* __restrict__ h1_acc, const float* __restrict__ h2_acc,
    const float* __restrict__ gh, const float* __restrict__ bh,
    float* __restrict__ out) {
  int tid = threadIdx.x, w = tid >> 5, lane = tid & 31;
  int R = blockIdx.x * 8 + w;  // 0..1023
  const float* accRow;
  const float* baseRow;
  float* oRow;
  if (R < 512) {
    accRow = h1_acc + (size_t)R * 256;
    baseRow = h1 + (size_t)R * 256;
    oRow = out + (size_t)R * 256;
  } else {
    int r = R - 512;
    accRow = h2_acc + (size_t)r * 256;
    baseRow = h2 + (size_t)r * 256;
    oRow = out + ROWS_H + (size_t)r * 256;
  }
  int h0 = lane * 8;
  float x[8];
  float s = 0.f, q = 0.f;
  #pragma unroll
  for (int j = 0; j < 8; ++j) {
    x[j] = accRow[h0 + j];
    s += x[j];
    q += x[j] * x[j];
  }
  for (int m = 16; m >= 1; m >>= 1) {
    s += __shfl_xor(s, m, 32);
    q += __shfl_xor(q, m, 32);
  }
  float mean = s * (1.f / 256.f);
  float var = q * (1.f / 256.f) - mean * mean;
  float rstd = rsqrtf(var + 1e-5f);
  #pragma unroll
  for (int j = 0; j < 8; ++j) {
    float y = (x[j] - mean) * rstd * gh[h0 + j] + bh[h0 + j];
    oRow[h0 + j] = baseRow[h0 + j] + fmaxf(y, 0.f);
  }
}

// ---------------------------------------------------------------------------
extern "C" void kernel_launch(void* const* d_in, const int* in_sizes, int n_in,
                              void* d_out, int out_size, void* d_ws,
                              size_t ws_size, hipStream_t stream) {
  (void)in_sizes; (void)n_in; (void)out_size; (void)ws_size;
  const float* h1 = (const float*)d_in[0];
  const float* h2 = (const float*)d_in[1];
  const float* e = (const float*)d_in[2];
  // d_in[3] = graph (unused by sum aggregation)
  const float* U1w = (const float*)d_in[4];  const float* U1b = (const float*)d_in[5];
  const float* V1w = (const float*)d_in[6];  const float* V1b = (const float*)d_in[7];
  const float* U2w = (const float*)d_in[8];  const float* U2b = (const float*)d_in[9];
  const float* V2w = (const float*)d_in[10]; const float* V2b = (const float*)d_in[11];
  const float* Aw  = (const float*)d_in[12]; const float* Ab  = (const float*)d_in[13];
  const float* Bmw = (const float*)d_in[14]; const float* Bmb = (const float*)d_in[15];
  const float* Cw  = (const float*)d_in[16]; const float* Cb  = (const float*)d_in[17];
  const float* gh = (const float*)d_in[18];  const float* bhh = (const float*)d_in[19];
  const float* ge = (const float*)d_in[20];  const float* be = (const float*)d_in[21];

  // workspace layout: 7x bf16-hi [n][k], 7x bf16-lo [n][k], then 6 f32 tensors
  unsigned short* wHiT = (unsigned short*)d_ws;
  unsigned short* wLoT = wHiT + (size_t)7 * 65536;
  float* fws = (float*)(wLoT + (size_t)7 * 65536);
  float* Ah = fws + (size_t)0 * ROWS_H;
  float* Bh = fws + (size_t)1 * ROWS_H;
  float* V1h = fws + (size_t)2 * ROWS_H;
  float* V2h = fws + (size_t)3 * ROWS_H;
  float* h1_acc = fws + (size_t)4 * ROWS_H;
  float* h2_acc = fws + (size_t)5 * ROWS_H;

  float* out = (float*)d_out;
  float* e_out = out + (size_t)2 * ROWS_H;

  // weight order in ws: U1, U2, V1, V2, A, Bm, C
  conv_w<<<dim3(4, 4, 7), 256, 0, stream>>>(U1w, U2w, V1w, V2w, Aw, Bmw, Cw,
                                            wHiT, wLoT);
  small_gemms<<<dim3(4, 6), 256, 0, stream>>>(h1, h2, wHiT, wLoT, U1b, U2b,
                                              V1b, V2b, Ab, Bmb, h1_acc,
                                              h2_acc, V1h, V2h, Ah, Bh);
  fused_e<<<dim3(1024), 256, 0, stream>>>(
      e, wHiT + (size_t)6 * 65536, wLoT + (size_t)6 * 65536, Cb, Ah, Bh, V1h,
      V2h, ge, be, h1_acc, h2_acc, e_out);
  finalize_h<<<dim3(128), 256, 0, stream>>>(h1, h2, h1_acc, h2_acc, gh, bhh,
                                            out);
}
